// SelfAttention_gate_72060961292412
// MI455X (gfx1250) — matrix operand order repaired and checked by reading, not compiled, not run on hardware
//
#include <hip/hip_runtime.h>
#include <math.h>

// ---------------- CDNA5 WMMA types ----------------
typedef __attribute__((ext_vector_type(16))) __bf16 v16bf;
typedef __attribute__((ext_vector_type(8)))  float  v8f;
typedef __attribute__((ext_vector_type(4)))  unsigned int u32x4;

#define N_TOT   16384
#define T_SEQ   6
#define C_DIM   1024
#define H_NUM   8
#define HD_DIM  128
#define SAMPLES 8
#define ROWS    (SAMPLES * T_SEQ)       // 48 rows per workgroup
#define A_STRIDE 1032                   // 1024 + 8 bf16 pad (bank-conflict-free frag loads)
#define QK_STRIDE 132                   // 128 + 4 f32 pad
#define LDS_BYTES (ROWS * A_STRIDE * 2 + 2 * ROWS * QK_STRIDE * 4 + (ROWS + SAMPLES * 36) * 4)

__device__ __forceinline__ unsigned short f2bf(float f) {
  unsigned int u = __float_as_uint(f);
  u += 0x7FFFu + ((u >> 16) & 1u);      // round-to-nearest-even
  return (unsigned short)(u >> 16);
}
__device__ __forceinline__ float bf2f(unsigned short h) {
  return __uint_as_float(((unsigned int)h) << 16);
}

union FragBF { u32x4 u[2]; v16bf b; };

// Transpose + convert weights: Wt[n][k] = (bf16) W[k][n]
__global__ void prep_weights_kernel(const float* __restrict__ Wk,
                                    const float* __restrict__ Wq,
                                    unsigned short* __restrict__ WtK,
                                    unsigned short* __restrict__ WtQ) {
  int idx = blockIdx.x * blockDim.x + threadIdx.x;  // 0 .. 1M-1
  int n = idx >> 10;
  int k = idx & 1023;
  WtK[idx] = f2bf(Wk[k * C_DIM + n]);
  WtQ[idx] = f2bf(Wq[k * C_DIM + n]);
}

__global__ __launch_bounds__(256) void fused_attn_kernel(
    const float* __restrict__ emb, const float* __restrict__ acv,
    const float* __restrict__ bk,  const float* __restrict__ bq,
    const float* __restrict__ Wg,  const float* __restrict__ bg,
    const unsigned short* __restrict__ WtK,
    const unsigned short* __restrict__ WtQ,
    float* __restrict__ ctx_out, float* __restrict__ attn_out) {
  extern __shared__ char smem[];
  unsigned short* sA   = (unsigned short*)smem;                    // 48 x 1032 bf16 (emb / V)
  float* sQ    = (float*)(smem + ROWS * A_STRIDE * 2);             // 48 x 132
  float* sK    = sQ + ROWS * QK_STRIDE;                            // 48 x 132
  float* sGate = sK + ROWS * QK_STRIDE;                            // 48
  float* sAttn = sGate + ROWS;                                     // 288

  const int tid    = threadIdx.x;
  const int lane   = tid & 31;
  const int wave   = tid >> 5;
  const int laneLo = lane & 15;
  const int laneHi = lane >> 4;
  const int n0     = blockIdx.x * SAMPLES;

  // ---- Phase 1: stage 48x1024 fp32 emb rows -> bf16 LDS (coalesced float4) ----
  const float* gsrc = emb + (size_t)n0 * T_SEQ * C_DIM;            // contiguous 49152 floats
  for (int v = tid; v < ROWS * C_DIM / 4; v += 256) {
    float4 f = ((const float4*)gsrc)[v];
    int e = v * 4;
    int r = e >> 10;
    int c = e & 1023;
    unsigned short* d = sA + r * A_STRIDE + c;
    d[0] = f2bf(f.x); d[1] = f2bf(f.y); d[2] = f2bf(f.z); d[3] = f2bf(f.w);
  }
  // ---- Gates: g[s][j] = ac[s] . Wg[:,j] + bg[j] ----
  if (tid < ROWS) {
    int s = tid / T_SEQ, j = tid % T_SEQ;
    const float* arow = acv + (size_t)(n0 + s) * 128;
    float g = bg[j];
    for (int a2 = 0; a2 < 128; ++a2) g += arow[a2] * Wg[a2 * 6 + j];
    sGate[tid] = g;
  }
  __syncthreads();

  const float invsq = 0.08838834764831845f;  // 1/sqrt(128)

  for (int h = 0; h < H_NUM; ++h) {
    // ---- Phase 2: Q/K projection for this head via bf16 WMMA ----
    v8f accQ0 = {}, accQ1 = {}, accQ2 = {};
    v8f accK0 = {}, accK1 = {}, accK2 = {};
    const int ncol = h * HD_DIM + wave * 16 + laneLo;      // output column owned by this lane
    const unsigned short* wqrow = WtQ + (size_t)ncol * C_DIM;
    const unsigned short* wkrow = WtK + (size_t)ncol * C_DIM;
    const unsigned short* abase = sA + laneLo * A_STRIDE;

    for (int ks = 0; ks < C_DIM / 32; ++ks) {
      const int k0 = ks * 32;
      // B frags: K chunks of 8 at (k0 + hi*8) and (k0 + 16 + hi*8)
      const int bc = k0 + laneHi * 8;
      FragBF fbq, fbk;
      fbq.u[0] = *(const u32x4*)(wqrow + bc);
      fbq.u[1] = *(const u32x4*)(wqrow + bc + 16);
      fbk.u[0] = *(const u32x4*)(wkrow + bc);
      fbk.u[1] = *(const u32x4*)(wkrow + bc + 16);
      // A frags: K chunks of 8 at (k0 + hi*8) and (k0 + 16 + hi*8), row = mt*16 + laneLo
      const int ac0 = k0 + laneHi * 8;
      FragBF fa0, fa1, fa2;
      fa0.u[0] = *(const u32x4*)(abase + ac0);
      fa0.u[1] = *(const u32x4*)(abase + ac0 + 16);
      fa1.u[0] = *(const u32x4*)(abase + 16 * A_STRIDE + ac0);
      fa1.u[1] = *(const u32x4*)(abase + 16 * A_STRIDE + ac0 + 16);
      fa2.u[0] = *(const u32x4*)(abase + 32 * A_STRIDE + ac0);
      fa2.u[1] = *(const u32x4*)(abase + 32 * A_STRIDE + ac0 + 16);

      accQ0 = __builtin_amdgcn_wmma_f32_16x16x32_bf16(false, fa0.b, false, fbq.b, (short)0, accQ0, false, false);
      accK0 = __builtin_amdgcn_wmma_f32_16x16x32_bf16(false, fa0.b, false, fbk.b, (short)0, accK0, false, false);
      accQ1 = __builtin_amdgcn_wmma_f32_16x16x32_bf16(false, fa1.b, false, fbq.b, (short)0, accQ1, false, false);
      accK1 = __builtin_amdgcn_wmma_f32_16x16x32_bf16(false, fa1.b, false, fbk.b, (short)0, accK1, false, false);
      accQ2 = __builtin_amdgcn_wmma_f32_16x16x32_bf16(false, fa2.b, false, fbq.b, (short)0, accQ2, false, false);
      accK2 = __builtin_amdgcn_wmma_f32_16x16x32_bf16(false, fa2.b, false, fbk.b, (short)0, accK2, false, false);
    }

    // ---- Spill Q/K (+bias) to LDS. C layout: VGPR v -> row v + laneHi*8, col laneLo ----
    {
      const int col = wave * 16 + laneLo;
      const float bqv = bq[h * HD_DIM + col];
      const float bkv = bk[h * HD_DIM + col];
#pragma unroll
      for (int v = 0; v < 8; ++v) {
        int r0 = laneHi * 8 + v;
        sQ[(r0     ) * QK_STRIDE + col] = accQ0[v] + bqv;
        sK[(r0     ) * QK_STRIDE + col] = accK0[v] + bkv;
        sQ[(r0 + 16) * QK_STRIDE + col] = accQ1[v] + bqv;
        sK[(r0 + 16) * QK_STRIDE + col] = accK1[v] + bkv;
        sQ[(r0 + 32) * QK_STRIDE + col] = accQ2[v] + bqv;
        sK[(r0 + 32) * QK_STRIDE + col] = accK2[v] + bkv;
      }
    }
    __syncthreads();

    // ---- Phase 3: raw scores (288 dot products of length 128) ----
    for (int p = tid; p < SAMPLES * 36; p += 256) {
      int s = p / 36, r = p % 36, q = r / 6, k = r % 6;
      const float* qp = sQ + (s * 6 + q) * QK_STRIDE;
      const float* kp = sK + (s * 6 + k) * QK_STRIDE;
      float dot = 0.f;
#pragma unroll 8
      for (int d2 = 0; d2 < HD_DIM; ++d2) dot += qp[d2] * kp[d2];
      sAttn[p] = dot;
    }
    __syncthreads();

    // ---- Phase 4: gated softmax over k; emit attention output ----
    if (tid < ROWS) {
      int s = tid / 6, q = tid % 6;
      float* row = sAttn + s * 36 + q * 6;
      float sc[6], m = -1e30f;
#pragma unroll
      for (int k = 0; k < 6; ++k) {
        sc[k] = row[k] * invsq * sGate[s * 6 + k];
        m = fmaxf(m, sc[k]);
      }
      float sum = 0.f;
#pragma unroll
      for (int k = 0; k < 6; ++k) { sc[k] = expf(sc[k] - m); sum += sc[k]; }
      float rs = 1.f / sum;
      float* aout = attn_out + (((size_t)(n0 + s) * H_NUM + h) * 36 + (size_t)q * 6);
#pragma unroll
      for (int k = 0; k < 6; ++k) {
        float a = sc[k] * rs;
        row[k] = a;
        aout[k] = a;
      }
    }
    __syncthreads();

    // ---- Phase 5: context = attn . V  (V = bf16 emb in LDS) ----
    for (int o = tid; o < SAMPLES * T_SEQ * HD_DIM; o += 256) {
      int s  = o / (T_SEQ * HD_DIM);
      int r  = o % (T_SEQ * HD_DIM);
      int q  = r / HD_DIM;
      int d2 = r % HD_DIM;
      const float* arow = sAttn + s * 36 + q * 6;
      const unsigned short* vbase = sA + (s * 6) * A_STRIDE + h * HD_DIM + d2;
      float cacc = 0.f;
#pragma unroll
      for (int k = 0; k < 6; ++k) cacc += arow[k] * bf2f(vbase[k * A_STRIDE]);
      ctx_out[((size_t)(n0 + s) * T_SEQ + q) * C_DIM + h * HD_DIM + d2] = cacc;
    }
    // No extra barrier needed: next head's WMMA phase only touches registers/sA,
    // and its sQ/sK/sAttn writes are ordered by the barriers above.
  }
}

extern "C" void kernel_launch(void* const* d_in, const int* in_sizes, int n_in,
                              void* d_out, int out_size, void* d_ws, size_t ws_size,
                              hipStream_t stream) {
  const float* emb = (const float*)d_in[0];
  const float* acv = (const float*)d_in[1];
  const float* Wk  = (const float*)d_in[2];
  const float* bk  = (const float*)d_in[3];
  const float* Wq  = (const float*)d_in[4];
  const float* bq  = (const float*)d_in[5];
  const float* Wg  = (const float*)d_in[6];
  const float* bg  = (const float*)d_in[7];

  float* ctx_out  = (float*)d_out;
  float* attn_out = ctx_out + (size_t)N_TOT * T_SEQ * C_DIM;

  unsigned short* WtK = (unsigned short*)d_ws;            // 1024x1024 bf16 (2 MB)
  unsigned short* WtQ = WtK + (size_t)C_DIM * C_DIM;      // 2 MB

  prep_weights_kernel<<<(C_DIM * C_DIM) / 256, 256, 0, stream>>>(Wk, Wq, WtK, WtQ);

  (void)hipFuncSetAttribute((const void*)fused_attn_kernel,
                            hipFuncAttributeMaxDynamicSharedMemorySize, LDS_BYTES);

  fused_attn_kernel<<<N_TOT / SAMPLES, 256, LDS_BYTES, stream>>>(
      emb, acv, bk, bq, Wg, bg, WtK, WtQ, ctx_out, attn_out);
}
